// DiffusionModel_87668872446339
// MI455X (gfx1250) — matrix operand order repaired and checked by reading, not compiled, not run on hardware
//
#include <hip/hip_runtime.h>
#include <hip/hip_bf16.h>

// ---------------------------------------------------------------------------
// Cross-attention forward for MI455X (gfx1250, wave32, WMMA + async-to-LDS).
//   Wt = transpose+bf16(W)           (tiny, once)
//   q  = X @ Wqt + bq  -> Qb  bf16 [B*SQ, D]
//   k  = C @ Wkt + bk  -> Kt  bf16 [B, D, SK]   (written transposed)
//   v  = C @ Wvt + bv  -> Vb  bf16 [B*SK, D]
//   S  = Qb @ Kt  (NN,f32) -> P = softmax(S) (bf16) -> out = P @ Vb (f32)
// NN WMMA GEMM: block tile 128x256x32, 8 waves, wave tile 32x128
// (16x v_wmma_f32_16x16x32_bf16 per wave per K-step), double-buffered LDS
// with global_load_async_to_lds_b128 staging (ASYNCcnt path) so tile t+1
// copies overlap tile t compute.
// ---------------------------------------------------------------------------

typedef __attribute__((ext_vector_type(16))) __bf16 v16bf;
typedef __attribute__((ext_vector_type(8)))  __bf16 v8bf;
typedef __attribute__((ext_vector_type(8)))  float  v8f;

union FragBF {
    v16bf v16;
    v8bf  h[2];
};

#define BM 128
#define BN 256
#define BK 32
#define LDS_A_PITCH 40    // bf16/row: 80B, 16B aligned, conflict-free strides
#define LDS_B_PITCH 264   // bf16/row: 528B, 16B aligned

// ---- async global->LDS copy helpers (gfx1250 ASYNCcnt path) ----
__device__ __forceinline__ void async_b128(unsigned ldsAddr, const void* g)
{
    asm volatile("global_load_async_to_lds_b128 %0, %1, off"
                 :: "v"(ldsAddr), "v"(g) : "memory");
}
__device__ __forceinline__ void async_b128_off16(unsigned ldsAddr, const void* g)
{
    asm volatile("global_load_async_to_lds_b128 %0, %1, off offset:16"
                 :: "v"(ldsAddr), "v"(g) : "memory");
}
__device__ __forceinline__ void wait_async0()
{
    asm volatile("s_wait_asynccnt 0x0" ::: "memory");
}

// ---- A-tile staging: 16 elements (one thread's share) ----
// f32 source: explicit b128 loads + convert + vector LDS store
__device__ __forceinline__ void stage_a16(const float* src, __bf16* dst)
{
    const float4* s4 = (const float4*)src;
    float4 f0 = s4[0], f1 = s4[1], f2 = s4[2], f3 = s4[3];
    v8bf lo, hi;
    lo[0] = (__bf16)f0.x; lo[1] = (__bf16)f0.y;
    lo[2] = (__bf16)f0.z; lo[3] = (__bf16)f0.w;
    lo[4] = (__bf16)f1.x; lo[5] = (__bf16)f1.y;
    lo[6] = (__bf16)f1.z; lo[7] = (__bf16)f1.w;
    hi[0] = (__bf16)f2.x; hi[1] = (__bf16)f2.y;
    hi[2] = (__bf16)f2.z; hi[3] = (__bf16)f2.w;
    hi[4] = (__bf16)f3.x; hi[5] = (__bf16)f3.y;
    hi[6] = (__bf16)f3.z; hi[7] = (__bf16)f3.w;
    *(v8bf*)(dst)     = lo;
    *(v8bf*)(dst + 8) = hi;
}
// bf16 source: async copy straight into LDS (no VGPR round-trip)
__device__ __forceinline__ void stage_a16(const __bf16* src, __bf16* dst)
{
    const unsigned l = (unsigned)(size_t)dst;   // low 32 bits = LDS offset
    async_b128(l, src);
    async_b128_off16(l, src);
}

// C = A(MxK, row-major) * B(KxN bf16, row-major) [+ bias]
// TRANS_C: store C transposed per batch: Ct[batch][col][row%mPerBatch]
template <typename AT, bool HAS_BIAS, bool TRANS_C, typename OutT>
__global__ __launch_bounds__(256)
void gemm_nn_wmma_kernel(const AT* __restrict__ Aall,
                         const __bf16* __restrict__ Ball,
                         const float* __restrict__ bias,
                         OutT* __restrict__ Call,
                         int M, int N, int K,
                         long batchStrideA, long batchStrideB, long batchStrideC,
                         int mPerBatch)
{
    __shared__ __bf16 ldsA[2][BM * LDS_A_PITCH];
    __shared__ __bf16 ldsB[2][BK * LDS_B_PITCH];

    const int z = blockIdx.z;
    const AT*     A = Aall + (size_t)z * batchStrideA;
    const __bf16* B = Ball + (size_t)z * batchStrideB;
    OutT*         C = Call + (size_t)z * batchStrideC;

    const int tid  = threadIdx.x;
    const int lane = tid & 31;
    const int wave = tid >> 5;        // 0..7
    const int wm   = wave >> 1;       // 0..3 -> 32-row strip
    const int wn   = wave & 1;        // 0..1 -> 128-col half
    const int m0   = blockIdx.x * BM;
    const int n0   = blockIdx.y * BN;

    v8f acc[2][8] = {};

    // staging coordinates
    const int a_row = tid >> 1;            // 0..127, 16 elems each
    const int a_k   = (tid & 1) * 16;      // 0 or 16
    const int b_k   = tid >> 3;            // 0..31, 32 elems each
    const int b_n   = (tid & 7) * 32;      // 0..224

    const AT*     srcA0 = A + (size_t)(m0 + a_row) * K + a_k;
    const __bf16* srcB0 = B + (size_t)b_k * N + (n0 + b_n);

    __bf16*  dstA[2]   = {&ldsA[0][a_row * LDS_A_PITCH + a_k],
                          &ldsA[1][a_row * LDS_A_PITCH + a_k]};
    unsigned dstB_l[2] = {(unsigned)(size_t)&ldsB[0][b_k * LDS_B_PITCH + b_n],
                          (unsigned)(size_t)&ldsB[1][b_k * LDS_B_PITCH + b_n]};

    auto stage_tile = [&](int k0, int buf) {
        stage_a16(srcA0 + k0, dstA[buf]);
        const __bf16* src = srcB0 + (size_t)k0 * N;
        const unsigned bl = dstB_l[buf];
        async_b128(bl, src);
        async_b128_off16(bl, src);
        async_b128(bl + 32, src + 16);
        async_b128_off16(bl + 32, src + 16);
    };

    // prologue: stage first tile into buffer 0
    stage_tile(0, 0);

    int cur = 0;
    for (int k0 = 0; k0 < K; k0 += BK, cur ^= 1) {
        wait_async0();        // our copies for buffer `cur` are done
        __syncthreads();      // everyone's copies done; prev reads retired
        if (k0 + BK < K)
            stage_tile(k0 + BK, cur ^ 1);   // overlap next copies w/ compute

        // ---- A fragments (two 16x32 strips per wave) ----
        FragBF af[2];
        const int kb = (lane >> 4) * 8;
#pragma unroll
        for (int i = 0; i < 2; ++i) {
            const __bf16* base =
                &ldsA[cur][(wm * 32 + i * 16 + (lane & 15)) * LDS_A_PITCH];
            af[i].h[0] = *(const v8bf*)(base + kb);
            af[i].h[1] = *(const v8bf*)(base + kb + 16);
        }
        // ---- B fragments (eight 32x16 tiles per wave): lane holds n=lane&15, k in table order ----
        const __bf16* bbase = &ldsB[cur][kb * LDS_B_PITCH + wn * 128 + (lane & 15)];
        FragBF bf[8];
#pragma unroll
        for (int j = 0; j < 8; ++j) {
#pragma unroll
            for (int s = 0; s < 8; ++s) {
                bf[j].h[0][s] = bbase[s * LDS_B_PITCH + j * 16];
                bf[j].h[1][s] = bbase[(16 + s) * LDS_B_PITCH + j * 16];
            }
        }
#pragma unroll
        for (int i = 0; i < 2; ++i)
#pragma unroll
            for (int j = 0; j < 8; ++j)
                acc[i][j] = __builtin_amdgcn_wmma_f32_16x16x32_bf16(
                    false, af[i].v16, false, bf[j].v16, (short)0, acc[i][j],
                    false, false);
    }

    // ---- epilogue: VGPR r, lane L -> (M = r + (L>=16?8:0), N = L&15) ----
    float bv_[8];
#pragma unroll
    for (int j = 0; j < 8; ++j)
        bv_[j] = HAS_BIAS ? bias[n0 + wn * 128 + j * 16 + (lane & 15)] : 0.f;
#pragma unroll
    for (int i = 0; i < 2; ++i) {
#pragma unroll
        for (int r = 0; r < 8; ++r) {
            const int row = m0 + wm * 32 + i * 16 + r + ((lane >> 4) << 3);
#pragma unroll
            for (int j = 0; j < 8; ++j) {
                const int   col = n0 + wn * 128 + j * 16 + (lane & 15);
                const float val = acc[i][j][r] + bv_[j];
                if (TRANS_C) {
                    const int batch  = row / mPerBatch;
                    const int within = row - batch * mPerBatch;
                    Call[(size_t)batch * N * mPerBatch +
                         (size_t)col * mPerBatch + within] = (OutT)val;
                } else {
                    C[(size_t)row * N + col] = (OutT)val;
                }
            }
        }
    }
}

// ---------------------------------------------------------------------------
// Weight transpose + convert: W f32 [R][Cc] -> Wt bf16 [Cc][R]
// ---------------------------------------------------------------------------
__global__ __launch_bounds__(256)
void transpose_convert_kernel(const float* __restrict__ W,
                              __bf16* __restrict__ Wt, int R, int Cc)
{
    __shared__ float t[32][33];
    const int tx = threadIdx.x, ty = threadIdx.y;     // (32, 8)
    const int c0 = blockIdx.x * 32, r0 = blockIdx.y * 32;
#pragma unroll
    for (int i = 0; i < 4; ++i)
        t[ty + i * 8][tx] = W[(size_t)(r0 + ty + i * 8) * Cc + (c0 + tx)];
    __syncthreads();
#pragma unroll
    for (int i = 0; i < 4; ++i)
        Wt[(size_t)(c0 + ty + i * 8) * R + (r0 + tx)] =
            (__bf16)t[tx][ty + i * 8];
}

// ---------------------------------------------------------------------------
// Row softmax: f32 scores -> bf16 probabilities. One block per row (2048 cols)
// ---------------------------------------------------------------------------
__global__ __launch_bounds__(256)
void softmax_rows_kernel(const float* __restrict__ S, __bf16* __restrict__ P,
                         int cols)
{
    __shared__ float red[256];
    const int row = blockIdx.x;
    const int tid = threadIdx.x;
    const float* s = S + (size_t)row * cols + tid * 8;
    __bf16*      p = P + (size_t)row * cols + tid * 8;

    const float4* s4 = (const float4*)s;
    float4 u0 = s4[0], u1 = s4[1];
    float v[8] = {u0.x, u0.y, u0.z, u0.w, u1.x, u1.y, u1.z, u1.w};

    float lmax = -1e30f;
#pragma unroll
    for (int i = 0; i < 8; ++i) lmax = fmaxf(lmax, v[i]);
    red[tid] = lmax;
    __syncthreads();
    for (int off = 128; off > 0; off >>= 1) {
        if (tid < off) red[tid] = fmaxf(red[tid], red[tid + off]);
        __syncthreads();
    }
    const float m = red[0];
    __syncthreads();

    float lsum = 0.f;
#pragma unroll
    for (int i = 0; i < 8; ++i) {
        v[i] = __expf(v[i] - m);
        lsum += v[i];
    }
    red[tid] = lsum;
    __syncthreads();
    for (int off = 128; off > 0; off >>= 1) {
        if (tid < off) red[tid] += red[tid + off];
        __syncthreads();
    }
    const float inv = 1.0f / red[0];

    v8bf o;
#pragma unroll
    for (int i = 0; i < 8; ++i) o[i] = (__bf16)(v[i] * inv);
    *(v8bf*)p = o;
}

// ---------------------------------------------------------------------------
// Host-side orchestration
// ---------------------------------------------------------------------------
extern "C" void kernel_launch(void* const* d_in, const int* in_sizes, int n_in,
                              void* d_out, int out_size, void* d_ws,
                              size_t ws_size, hipStream_t stream)
{
    (void)in_sizes; (void)n_in; (void)out_size; (void)ws_size;

    constexpr int Bn = 4, SQ = 2048, SK = 2048, D = 1024;
    constexpr int MQ = Bn * SQ;   // 8192
    constexpr int MK = Bn * SK;   // 8192

    const float* X  = (const float*)d_in[0];
    const float* Cd = (const float*)d_in[1];
    const float* Wq = (const float*)d_in[2];
    const float* bq = (const float*)d_in[3];
    const float* Wk = (const float*)d_in[4];
    const float* bk = (const float*)d_in[5];
    const float* Wv = (const float*)d_in[6];
    const float* bv = (const float*)d_in[7];
    float* out = (float*)d_out;

    // workspace layout (L2-resident):
    // Wqt|Wkt|Wvt [D*D bf16]  Qb [MQ*D]  Kt [Bn*D*SK]  Vb [MK*D]
    // S [Bn*SQ*SK f32]  P [Bn*SQ*SK bf16]
    char* ws = (char*)d_ws;
    __bf16* Wqt = (__bf16*)ws;
    __bf16* Wkt = Wqt + (size_t)D * D;
    __bf16* Wvt = Wkt + (size_t)D * D;
    __bf16* Qb  = Wvt + (size_t)D * D;
    __bf16* Kt  = Qb + (size_t)MQ * D;
    __bf16* Vb  = Kt + (size_t)Bn * D * SK;
    float*  S   = (float*)(Vb + (size_t)MK * D);
    __bf16* P   = (__bf16*)(S + (size_t)Bn * SQ * SK);

    const dim3 blk(256);

    // --- weight transpose + bf16 convert (tiny) ---
    {
        dim3 tb(32, 8);
        dim3 tg(D / 32, D / 32);
        transpose_convert_kernel<<<tg, tb, 0, stream>>>(Wq, Wqt, D, D);
        transpose_convert_kernel<<<tg, tb, 0, stream>>>(Wk, Wkt, D, D);
        transpose_convert_kernel<<<tg, tb, 0, stream>>>(Wv, Wvt, D, D);
    }

    // --- projections (NN, f32 A, bf16 B) ---
    {
        dim3 grid(MQ / BM, D / BN, 1);
        gemm_nn_wmma_kernel<float, true, false, __bf16>
            <<<grid, blk, 0, stream>>>(X, Wqt, bq, Qb, MQ, D, D, 0, 0, 0, SQ);
        gemm_nn_wmma_kernel<float, true, true, __bf16>   // K written transposed
            <<<grid, blk, 0, stream>>>(Cd, Wkt, bk, Kt, MK, D, D, 0, 0, 0, SK);
        gemm_nn_wmma_kernel<float, true, false, __bf16>
            <<<grid, blk, 0, stream>>>(Cd, Wvt, bv, Vb, MK, D, D, 0, 0, 0, SK);
    }

    // --- scores: S[b] = Qb[b] @ Kt[b]  (NN), f32 out ---
    {
        dim3 grid(SQ / BM, SK / BN, Bn);
        gemm_nn_wmma_kernel<__bf16, false, false, float>
            <<<grid, blk, 0, stream>>>(Qb, Kt, nullptr, S, SQ, SK, D,
                                       (long)SQ * D, (long)D * SK,
                                       (long)SQ * SK, SQ);
    }

    // --- softmax rows -> bf16 probabilities ---
    softmax_rows_kernel<<<dim3(Bn * SQ), blk, 0, stream>>>(S, P, SK);

    // --- output: out[b] = P[b] @ Vb[b]  (NN), f32 out ---
    {
        dim3 grid(SQ / BM, D / BN, Bn);
        gemm_nn_wmma_kernel<__bf16, false, false, float>
            <<<grid, blk, 0, stream>>>(P, Vb, nullptr, out, SQ, D, SK,
                                       (long)SQ * SK, (long)SK * D,
                                       (long)SQ * D, SQ);
    }
}
